// SeparationHead_670014898682
// MI455X (gfx1250) — matrix operand order repaired and checked by reading, not compiled, not run on hardware
//
#include <hip/hip_runtime.h>
#include <hip/hip_bf16.h>
#include <stdint.h>

// ---------------------------------------------------------------------------
// SeparationHead for MI455X (gfx1250, wave32, WMMA).
//
// Shapes: B=1024, A=100, K=16, N=65536, D=128, G=128, H=128.
// Phase 1 (gather/mean) is L2-bound: node_features is 32MB << 192MB L2.
// Phase 2 (102400x256x128 GEMM) uses v_wmma_f32_16x16x32_f16.
// ---------------------------------------------------------------------------

typedef __attribute__((ext_vector_type(16))) _Float16 v16h;
typedef __attribute__((ext_vector_type(8)))  float    v8f;

#define Bsz 1024
#define Asz 100
#define Ksz 16
#define Dsz 128
#define Gsz 128
#define Hsz 128
#define IN_DIM 256           // G + D
#define A_TILE 16
#define N_A_TILES 7          // ceil(100/16)
#define NEGVAL (-1.0e9f)

// LDS row stride for the f16 input tile: 264 halves = 528 bytes (16B multiple,
// rows rotate 4 banks -> conflict-light ds_load_b128 fragment reads).
#define INP_STRIDE 264

// ---------------------------------------------------------------------------
// Prologue: pack W1 [256 x 128] f32 row-major into the exact per-lane VGPR
// image of the WMMA B fragments (f16, 2 values packed per dword).
//
// Fragment (32x16 f16 B, wave32), this code's contiguous k order (A loads match):
//   lane  0..15 : column n = lane,    VGPR v holds K = kbase+2v,   kbase+2v+1
//   lane 16..31 : column n = lane-16, VGPR v holds K = kbase+16+2v, +1
// Buffer layout: frag[((t*8 + s)*32 + lane)*8 + v]  (t = n-tile, s = k-step)
// -> each lane's 8 dwords are contiguous (two coalesced global_load_b128).
// Total: 8*8*32*8 dwords = 64 KB, L2-resident for the whole main kernel.
// ---------------------------------------------------------------------------
__global__ void w1_pack_kernel(const float* __restrict__ W1,
                               uint32_t* __restrict__ frag) {
  int idx  = blockIdx.x * 256 + threadIdx.x;     // 0..16383
  int t    = idx >> 11;                          // n-tile 0..7
  int s    = (idx >> 8) & 7;                     // k-step 0..7
  int lane = (idx >> 3) & 31;
  int v    = idx & 7;
  int n    = t * 16 + (lane & 15);
  int k    = s * 32 + ((lane >= 16) ? 16 : 0) + 2 * v;
  union { _Float16 h[2]; uint32_t u; } pk;
  pk.h[0] = (_Float16)W1[(size_t)k * Hsz + n];
  pk.h[1] = (_Float16)W1[(size_t)(k + 1) * Hsz + n];
  frag[idx] = pk.u;
}

// ---------------------------------------------------------------------------
// Main kernel: grid (B, 7), block 256 (8 waves).
// ---------------------------------------------------------------------------
__global__ __launch_bounds__(256)
void sep_head_kernel(const float* __restrict__ nf,        // [N, D]
                     const float* __restrict__ gf,        // [B, G]
                     const int*   __restrict__ cube_mask, // [N]
                     const int*   __restrict__ sep_rows,  // [B, A, K]
                     const int*   __restrict__ sep_valid, // [B, A, K]
                     const int*   __restrict__ sep_mask,  // [B, A]
                     const uint32_t* __restrict__ w1frag, // packed B frags
                     const float* __restrict__ b1,        // [H]
                     const float* __restrict__ w2,        // [H]
                     const float* __restrict__ b2,        // [1]
                     float*       __restrict__ out) {     // [B, A]
  const int bIdx = blockIdx.x;
  const int a0   = blockIdx.y * A_TILE;

  __shared__ alignas(16) _Float16 inp[A_TILE][INP_STRIDE]; // f16 [16 x 256]
  __shared__ float score[A_TILE];

  const int tid = threadIdx.x;

  // ---------------- Phase 1: masked-mean gather + build f16 input tile -----
  {
    const int a_local = tid >> 4;      // 0..15 : row of A tile
    const int lane16  = tid & 15;      // 16 lanes per row, 8 channels each
    const int a       = a0 + a_local;

    float acc[8];
#pragma unroll
    for (int j = 0; j < 8; ++j) acc[j] = 0.0f;
    float cnt = 0.0f;

    if (a < Asz) {
      const int base = (bIdx * Asz + a) * Ksz;
#pragma unroll 1
      for (int k = 0; k < Ksz; ++k) {
        if (sep_valid[base + k]) {
          cnt += 1.0f;
          const int   row = sep_rows[base + k];
          const float msk = (float)cube_mask[row];   // reference cube zeroing
          const float* f  = nf + (size_t)row * Dsz + lane16;
#pragma unroll
          for (int j = 0; j < 8; ++j) acc[j] += msk * f[j * 16];
        }
      }
    }
    const float inv = (cnt > 0.0f) ? (1.0f / cnt) : 0.0f;
    const float* g  = gf + (size_t)bIdx * Gsz;
#pragma unroll
    for (int j = 0; j < 8; ++j) {
      const int d = lane16 + j * 16;
      inp[a_local][d]       = (_Float16)g[d];           // global part
      inp[a_local][Dsz + d] = (_Float16)(acc[j] * inv); // sep_emb part
    }
  }
  if (tid < A_TILE) score[tid] = 0.0f;
  __syncthreads();

  // ---------------- Phase 2: 16x256 @ 256x128 via WMMA ---------------------
  const int wave = tid >> 5;           // n-tile t = wave (H columns t*16..)
  const int lane = tid & 31;
  const int mrow = lane & 15;          // A-frag row / B-frag column index
  const int koff = (lane < 16) ? 0 : 16;

  v8f acc8 = {};
  const uint4* bbase =
      (const uint4*)w1frag + ((size_t)wave * 8) * 64 + lane * 2;

#pragma unroll
  for (int s = 0; s < 8; ++s) {
    const int kb = s * 32;
    union { v16h v; uint4 q[2]; } Af, Bf;
    // A fragment from LDS (16B-aligned ds_load_b128 x2):
    //   lanes 0-15 : K = kb+0..15
    //   lanes16-31 : K = kb+16..31
    Af.q[0] = *(const uint4*)&inp[mrow][kb + koff];
    Af.q[1] = *(const uint4*)&inp[mrow][kb + koff + 8];
    // B fragment from packed global buffer (32 contiguous bytes per lane).
    const uint4* bp = bbase + (size_t)s * 64;
    Bf.q[0] = bp[0];
    Bf.q[1] = bp[1];
    acc8 = __builtin_amdgcn_wmma_f32_16x16x32_f16(
        /*neg_a=*/false, Af.v, /*neg_b=*/false, Bf.v,
        /*c_mod=*/(short)0, acc8, /*reuse_a=*/false, /*reuse_b=*/false);
  }

  // ---------------- Phase 3: bias + ReLU + dot(W2) + reduce ----------------
  const int   n   = wave * 16 + mrow;  // this lane's H column
  const float b1n = b1[n];
  const float w2n = w2[n];

#pragma unroll
  for (int r = 0; r < 8; ++r) {
    float h = acc8[r] + b1n;
    h = h > 0.0f ? h : 0.0f;
    float p = h * w2n;
    // Sum the 16 H-columns held by each 16-lane half (C/D layout:
    // lanes 0-15 -> row r, lanes 16-31 -> row r+8).
    p += __shfl_xor(p, 1, 32);
    p += __shfl_xor(p, 2, 32);
    p += __shfl_xor(p, 4, 32);
    p += __shfl_xor(p, 8, 32);
    if ((lane & 15) == 0) {
      const int m = (lane < 16) ? r : (r + 8);
      atomicAdd(&score[m], p);   // combine 8 waves' H-tiles
    }
  }
  __syncthreads();

  if (tid < A_TILE) {
    const int a = a0 + tid;
    if (a < Asz) {
      const int idx = bIdx * Asz + a;
      out[idx] = sep_mask[idx] ? (score[tid] + b2[0]) : NEGVAL;
    }
  }
}

// ---------------------------------------------------------------------------
// Host entry.  d_in order (setup_inputs dict):
//  0 node_features f32 | 1 global_features f32 | 2 cube_mask int | 3 batch int
//  4 sep_rows int | 5 sep_valid int | 6 sep_mask int
//  7 W1 f32 | 8 b1 f32 | 9 W2 f32 | 10 b2 f32
// ---------------------------------------------------------------------------
extern "C" void kernel_launch(void* const* d_in, const int* in_sizes, int n_in,
                              void* d_out, int out_size, void* d_ws,
                              size_t ws_size, hipStream_t stream) {
  const float* nf        = (const float*)d_in[0];
  const float* gf        = (const float*)d_in[1];
  const int*   cube_mask = (const int*)d_in[2];
  const int*   sep_rows  = (const int*)d_in[4];
  const int*   sep_valid = (const int*)d_in[5];
  const int*   sep_mask  = (const int*)d_in[6];
  const float* W1        = (const float*)d_in[7];
  const float* b1        = (const float*)d_in[8];
  const float* W2        = (const float*)d_in[9];
  const float* b2        = (const float*)d_in[10];
  float*       out       = (float*)d_out;

  uint32_t* w1frag = (uint32_t*)d_ws;  // 64 KB packed B fragments

  // Pack W1 into WMMA B-fragment layout (16384 dwords).
  w1_pack_kernel<<<64, 256, 0, stream>>>(W1, w1frag);

  // Main fused kernel: 1024 batches x 7 A-tiles, 8 waves each.
  sep_head_kernel<<<dim3(Bsz, N_A_TILES), 256, 0, stream>>>(
      nf, gf, cube_mask, sep_rows, sep_valid, sep_mask, w1frag, b1, W2, b2,
      out);
}
